// CausalSelfAttention_66486093742084
// MI455X (gfx1250) — matrix operand order repaired and checked by reading, not compiled, not run on hardware
//
#include <hip/hip_runtime.h>
#include <math.h>

#define D_MODEL 1024
#define T_SEQ   2048
#define BATCH   4
#define HEADS   16
#define HDIM    64

typedef __attribute__((ext_vector_type(16))) __bf16       bf16x16;
typedef __attribute__((ext_vector_type(8)))  float        f32x8;
typedef __attribute__((ext_vector_type(4)))  unsigned int u32x4;

__device__ __forceinline__ unsigned short f32_to_bf16(float f) {
  union { float f; unsigned u; } v; v.f = f;
  unsigned r = v.u + 0x7FFFu + ((v.u >> 16) & 1u);   // round-to-nearest-even
  return (unsigned short)(r >> 16);
}

struct Frag2 { u32x4 a, b; };
__device__ __forceinline__ bf16x16 frag_ld(const unsigned short* p0, const unsigned short* p1) {
  union { Frag2 s; bf16x16 v; } u;
  u.s.a = *reinterpret_cast<const u32x4*>(p0);
  u.s.b = *reinterpret_cast<const u32x4*>(p1);
  return u.v;
}

// CDNA5 async global->LDS copy, 16B per lane, tracked with ASYNCcnt.
__device__ __forceinline__ void async_ld16(unsigned short* lds_ptr,
                                           const unsigned short* gptr) {
  unsigned lds_off = (unsigned)(size_t)lds_ptr;  // addr[31:0] = LDS byte offset
  asm volatile("global_load_async_to_lds_b128 %0, %1, off"
               :: "v"(lds_off), "v"(gptr)
               : "memory");
}

// ---------------- conversion kernels ----------------
__global__ void k_cvt(const float* __restrict__ in, unsigned short* __restrict__ out, int n) {
  int i = blockIdx.x * blockDim.x + threadIdx.x;
  int stride = gridDim.x * blockDim.x;
  for (; i < n; i += stride) out[i] = f32_to_bf16(in[i]);
}

// in: [K][N] row-major fp32  ->  out: [N][K] bf16 (pre-transposed weights)
__global__ void k_cvt_t(const float* __restrict__ in, unsigned short* __restrict__ out,
                        int K, int N) {
  int i = blockIdx.x * blockDim.x + threadIdx.x;
  int stride = gridDim.x * blockDim.x;
  int total = K * N;
  for (; i < total; i += stride) {
    int n = i / K, k = i - n * K;
    out[i] = f32_to_bf16(in[(size_t)k * N + n]);
  }
}

// Issue one 128x64 bf16 tile of A and of Bt into LDS (8 async ops/thread).
__device__ __forceinline__ void issue_tiles(
    unsigned short* dstA, unsigned short* dstB,
    const unsigned short* __restrict__ A, const unsigned short* __restrict__ Bt,
    int m0, int n0, int k0, int tid) {
#pragma unroll
  for (int i = 0; i < 4; ++i) {
    int c = tid + i * 256;          // 1024 chunks of 16B per 16KB tile
    int row = c >> 3, e = (c & 7) * 8;
    async_ld16(dstA + row * 64 + e, A  + (size_t)(m0 + row) * 1024 + k0 + e);
    async_ld16(dstB + row * 64 + e, Bt + (size_t)(n0 + row) * 1024 + k0 + e);
  }
}

// ---------------- GEMM 1: qkv = x @ W_qkv + b_qkv, scatter to Q/K/Vt ----------------
__global__ __launch_bounds__(256) void k_gemm_qkv(
    const unsigned short* __restrict__ A,    // x bf16 [8192][1024]
    const unsigned short* __restrict__ Bt,   // W_qkv^T bf16 [3072][1024]
    const float* __restrict__ bias,
    unsigned short* __restrict__ Qb,         // [B,H,T,64]
    unsigned short* __restrict__ Kb,         // [B,H,T,64]
    unsigned short* __restrict__ Vtb) {      // [B,H,64,T]
  __shared__ __align__(16) unsigned short As[2][128 * 64];  // 2 x 16KB
  __shared__ __align__(16) unsigned short Bs[2][128 * 64];  // 2 x 16KB
  const int K = 1024;
  int n0 = blockIdx.x * 128, m0 = blockIdx.y * 128;
  int tid = threadIdx.x;
  int wave = tid >> 5, lane = tid & 31, l16 = lane & 15, hl = lane >> 4;
  int wm = wave & 3, wn = wave >> 2;              // 4 (M) x 2 (N) wave grid
  f32x8 acc[2][4] = {};

  issue_tiles(As[0], Bs[0], A, Bt, m0, n0, 0, tid);
  for (int k0 = 0; k0 < K; k0 += 64) {
    int cur = (k0 >> 6) & 1;
    if (k0 + 64 < K) {
      issue_tiles(As[cur ^ 1], Bs[cur ^ 1], A, Bt, m0, n0, k0 + 64, tid);
      asm volatile("s_wait_asynccnt 0x8" ::: "memory");  // retire previous tile's 8 ops
    } else {
      asm volatile("s_wait_asynccnt 0x0" ::: "memory");
    }
    __syncthreads();
#pragma unroll
    for (int kk = 0; kk < 2; ++kk) {
      bf16x16 af[2], bfr[4];
#pragma unroll
      for (int mi = 0; mi < 2; ++mi) {   // A frag: lane=row, K at hl*8 / hl*8+16
        const unsigned short* p = As[cur] + (wm * 32 + mi * 16 + l16) * 64 + kk * 32 + hl * 8;
        af[mi] = frag_ld(p, p + 16);
      }
#pragma unroll
      for (int ni = 0; ni < 4; ++ni) {   // B frag: lane=col, K at hl*8 / hl*8+16
        const unsigned short* p = Bs[cur] + (wn * 64 + ni * 16 + l16) * 64 + kk * 32 + hl * 8;
        bfr[ni] = frag_ld(p, p + 16);
      }
#pragma unroll
      for (int mi = 0; mi < 2; ++mi)
#pragma unroll
        for (int ni = 0; ni < 4; ++ni)
          acc[mi][ni] = __builtin_amdgcn_wmma_f32_16x16x32_bf16(
              false, af[mi], false, bfr[ni], (short)0, acc[mi][ni], false, false);
    }
    __syncthreads();   // reads done before next iteration overwrites this buffer
  }

  // epilogue: bias + bf16 + scatter to Q / K / V^T
#pragma unroll
  for (int mi = 0; mi < 2; ++mi) {
#pragma unroll
    for (int ni = 0; ni < 4; ++ni) {
      int n = n0 + wn * 64 + ni * 16 + l16;       // 0..3071
      int which = n >> 10, c = n & 1023;
      int hh = c >> 6, d = c & 63;
      float bv = bias[n];
#pragma unroll
      for (int j = 0; j < 8; ++j) {
        int r = m0 + wm * 32 + mi * 16 + hl * 8 + j;  // token row 0..8191
        int bb = r >> 11, tt = r & 2047;
        unsigned short val = f32_to_bf16(acc[mi][ni][j] + bv);
        if (which == 0)
          Qb[(((size_t)(bb * HEADS + hh)) * T_SEQ + tt) * HDIM + d] = val;
        else if (which == 1)
          Kb[(((size_t)(bb * HEADS + hh)) * T_SEQ + tt) * HDIM + d] = val;
        else
          Vtb[(((size_t)(bb * HEADS + hh)) * HDIM + d) * T_SEQ + tt] = val;
      }
    }
  }
}

// ---------------- Flash attention (causal), one wave = 16 query rows ----------------
__global__ __launch_bounds__(256) void k_attn(
    const unsigned short* __restrict__ Q,   // [B,H,T,64]
    const unsigned short* __restrict__ Km,  // [B,H,T,64]
    const unsigned short* __restrict__ Vt,  // [B,H,64,T]
    unsigned short* __restrict__ AO) {      // [B,T,C] bf16
  __shared__ __align__(16) unsigned short Pbuf[8][16 * 32];
  int h = blockIdx.y, b = blockIdx.z;
  int wave = threadIdx.x >> 5, lane = threadIdx.x & 31;
  int l16 = lane & 15, hl = lane >> 4;
  const unsigned short* Qh = Q  + ((size_t)(b * HEADS + h)) * T_SEQ * HDIM;
  const unsigned short* Kh = Km + ((size_t)(b * HEADS + h)) * T_SEQ * HDIM;
  const unsigned short* Vh = Vt + ((size_t)(b * HEADS + h)) * HDIM * T_SEQ;
  int qr = blockIdx.x * 128 + wave * 16;

  bf16x16 qf[2];
#pragma unroll
  for (int dh = 0; dh < 2; ++dh) {
    const unsigned short* p = Qh + (size_t)(qr + l16) * HDIM + dh * 32 + hl * 8;
    qf[dh] = frag_ld(p, p + 16);
  }

  f32x8 o[4] = {};
  float mrow[8], lrow[8];
#pragma unroll
  for (int j = 0; j < 8; ++j) { mrow[j] = -1e30f; lrow[j] = 0.f; }
  unsigned short* pb = Pbuf[wave];

  int kend = qr + 16;
  for (int kc = 0; kc < kend; kc += 32) {
    f32x8 s0 = {}, s1 = {};
#pragma unroll
    for (int dh = 0; dh < 2; ++dh) {
      const unsigned short* p0 = Kh + (size_t)(kc + l16) * HDIM + dh * 32 + hl * 8;
      bf16x16 kf0 = frag_ld(p0, p0 + 16);
      s0 = __builtin_amdgcn_wmma_f32_16x16x32_bf16(false, qf[dh], false, kf0, (short)0, s0, false, false);
      const unsigned short* p1 = Kh + (size_t)(kc + 16 + l16) * HDIM + dh * 32 + hl * 8;
      bf16x16 kf1 = frag_ld(p1, p1 + 16);
      s1 = __builtin_amdgcn_wmma_f32_16x16x32_bf16(false, qf[dh], false, kf1, (short)0, s1, false, false);
    }
    // scale 1/sqrt(64) + causal mask (lane = column, VGPR j = row)
#pragma unroll
    for (int j = 0; j < 8; ++j) {
      int row = qr + hl * 8 + j;
      float v0 = s0[j] * 0.125f, v1 = s1[j] * 0.125f;
      if (kc + l16 > row)      v0 = -1e30f;
      if (kc + 16 + l16 > row) v1 = -1e30f;
      s0[j] = v0; s1[j] = v1;
    }
    // online softmax: reductions across the 16-lane half holding this row set
#pragma unroll
    for (int j = 0; j < 8; ++j) {
      float cm = fmaxf(s0[j], s1[j]);
      cm = fmaxf(cm, __shfl_xor(cm, 1, 32));
      cm = fmaxf(cm, __shfl_xor(cm, 2, 32));
      cm = fmaxf(cm, __shfl_xor(cm, 4, 32));
      cm = fmaxf(cm, __shfl_xor(cm, 8, 32));
      float mn = fmaxf(mrow[j], cm);
      float alpha = __expf(mrow[j] - mn);
      float p0 = __expf(s0[j] - mn);
      float p1 = __expf(s1[j] - mn);
      s0[j] = p0; s1[j] = p1;
      float rs = p0 + p1;
      rs += __shfl_xor(rs, 1, 32);
      rs += __shfl_xor(rs, 2, 32);
      rs += __shfl_xor(rs, 4, 32);
      rs += __shfl_xor(rs, 8, 32);
      lrow[j] = lrow[j] * alpha + rs;
      mrow[j] = mn;
#pragma unroll
      for (int dt = 0; dt < 4; ++dt) o[dt][j] *= alpha;
    }
    // P tile: C-layout regs -> LDS -> A-layout frag
#pragma unroll
    for (int j = 0; j < 8; ++j) {
      int row = hl * 8 + j;
      pb[row * 32 + l16]      = f32_to_bf16(s0[j]);
      pb[row * 32 + 16 + l16] = f32_to_bf16(s1[j]);
    }
    __asm__ volatile("s_wait_dscnt 0x0" ::: "memory");
    const unsigned short* pp = pb + l16 * 32 + hl * 8;
    bf16x16 pf = frag_ld(pp, pp + 16);
#pragma unroll
    for (int dt = 0; dt < 4; ++dt) {               // O += P @ V  (V cols from V^T rows)
      int d = dt * 16 + l16;
      const unsigned short* vp = Vh + (size_t)d * T_SEQ + kc + hl * 8;
      bf16x16 vf = frag_ld(vp, vp + 16);
      o[dt] = __builtin_amdgcn_wmma_f32_16x16x32_bf16(false, pf, false, vf, (short)0, o[dt], false, false);
    }
  }
  // normalize + store [B,T,C]
#pragma unroll
  for (int j = 0; j < 8; ++j) {
    float inv = 1.0f / lrow[j];
    int row = qr + hl * 8 + j;
    size_t base = ((size_t)b * T_SEQ + row) * D_MODEL + h * HDIM;
#pragma unroll
    for (int dt = 0; dt < 4; ++dt)
      AO[base + dt * 16 + l16] = f32_to_bf16(o[dt][j] * inv);
  }
}

// ---------------- GEMM 2: out = attn @ W_out + b_out (fp32 out) ----------------
__global__ __launch_bounds__(256) void k_gemm_out(
    const unsigned short* __restrict__ A,    // attn bf16 [8192][1024]
    const unsigned short* __restrict__ Bt,   // W_out^T bf16 [1024][1024]
    const float* __restrict__ bias,          // [1024]
    float* __restrict__ Out) {               // [8192][1024] fp32
  __shared__ __align__(16) unsigned short As[2][128 * 64];
  __shared__ __align__(16) unsigned short Bs[2][128 * 64];
  const int K = 1024;
  int n0 = blockIdx.x * 128, m0 = blockIdx.y * 128;
  int tid = threadIdx.x;
  int wave = tid >> 5, lane = tid & 31, l16 = lane & 15, hl = lane >> 4;
  int wm = wave & 3, wn = wave >> 2;
  f32x8 acc[2][4] = {};

  issue_tiles(As[0], Bs[0], A, Bt, m0, n0, 0, tid);
  for (int k0 = 0; k0 < K; k0 += 64) {
    int cur = (k0 >> 6) & 1;
    if (k0 + 64 < K) {
      issue_tiles(As[cur ^ 1], Bs[cur ^ 1], A, Bt, m0, n0, k0 + 64, tid);
      asm volatile("s_wait_asynccnt 0x8" ::: "memory");
    } else {
      asm volatile("s_wait_asynccnt 0x0" ::: "memory");
    }
    __syncthreads();
#pragma unroll
    for (int kk = 0; kk < 2; ++kk) {
      bf16x16 af[2], bfr[4];
#pragma unroll
      for (int mi = 0; mi < 2; ++mi) {
        const unsigned short* p = As[cur] + (wm * 32 + mi * 16 + l16) * 64 + kk * 32 + hl * 8;
        af[mi] = frag_ld(p, p + 16);
      }
#pragma unroll
      for (int ni = 0; ni < 4; ++ni) {
        const unsigned short* p = Bs[cur] + (wn * 64 + ni * 16 + l16) * 64 + kk * 32 + hl * 8;
        bfr[ni] = frag_ld(p, p + 16);
      }
#pragma unroll
      for (int mi = 0; mi < 2; ++mi)
#pragma unroll
        for (int ni = 0; ni < 4; ++ni)
          acc[mi][ni] = __builtin_amdgcn_wmma_f32_16x16x32_bf16(
              false, af[mi], false, bfr[ni], (short)0, acc[mi][ni], false, false);
    }
    __syncthreads();
  }
#pragma unroll
  for (int mi = 0; mi < 2; ++mi) {
#pragma unroll
    for (int ni = 0; ni < 4; ++ni) {
      int n = n0 + wn * 64 + ni * 16 + l16;
      float bv = bias[n];
#pragma unroll
      for (int j = 0; j < 8; ++j) {
        int r = m0 + wm * 32 + mi * 16 + hl * 8 + j;
        Out[(size_t)r * 1024 + n] = acc[mi][ni][j] + bv;
      }
    }
  }
}

// ---------------- launch ----------------
extern "C" void kernel_launch(void* const* d_in, const int* in_sizes, int n_in,
                              void* d_out, int out_size, void* d_ws, size_t ws_size,
                              hipStream_t stream) {
  (void)in_sizes; (void)n_in; (void)out_size; (void)ws_size;
  const float* x    = (const float*)d_in[0];
  const float* Wqkv = (const float*)d_in[1];
  const float* bqkv = (const float*)d_in[2];
  const float* Wout = (const float*)d_in[3];
  const float* bout = (const float*)d_in[4];
  float* out = (float*)d_out;

  char* ws = (char*)d_ws;
  size_t off = 0;
  auto carve = [&](size_t bytes) {
    void* p = ws + off;
    off += (bytes + 255) & ~(size_t)255;
    return p;
  };
  const size_t TOK = (size_t)BATCH * T_SEQ;                 // 8192
  unsigned short* xb  = (unsigned short*)carve(TOK * D_MODEL * 2);                 // 16 MB
  unsigned short* WqT = (unsigned short*)carve((size_t)3 * D_MODEL * D_MODEL * 2); // 6 MB
  unsigned short* WoT = (unsigned short*)carve((size_t)D_MODEL * D_MODEL * 2);     // 2 MB
  unsigned short* Qb  = (unsigned short*)carve(TOK * D_MODEL * 2);                 // 16 MB
  unsigned short* Kb  = (unsigned short*)carve(TOK * D_MODEL * 2);                 // 16 MB
  unsigned short* Vtb = (unsigned short*)carve(TOK * D_MODEL * 2);                 // 16 MB
  unsigned short* AO  = xb;  // x is dead after GEMM1; reuse for attention output

  k_cvt  <<<2048, 256, 0, stream>>>(x, xb, (int)(TOK * D_MODEL));
  k_cvt_t<<<2048, 256, 0, stream>>>(Wqkv, WqT, D_MODEL, 3 * D_MODEL);
  k_cvt_t<<<1024, 256, 0, stream>>>(Wout, WoT, D_MODEL, D_MODEL);

  k_gemm_qkv<<<dim3(3 * D_MODEL / 128, TOK / 128), 256, 0, stream>>>(
      xb, WqT, bqkv, Qb, Kb, Vtb);

  k_attn<<<dim3(T_SEQ / 128, HEADS, BATCH), 256, 0, stream>>>(Qb, Kb, Vtb, AO);

  k_gemm_out<<<dim3(D_MODEL / 128, TOK / 128), 256, 0, stream>>>(
      AO, WoT, bout, out);
}
